// SelfAttention_13219909337103
// MI455X (gfx1250) — compile-verified, matrix operand order checked by reading, not run on hardware
//
#include <hip/hip_runtime.h>

typedef _Float16 v16h __attribute__((ext_vector_type(16)));
typedef _Float16 h8   __attribute__((ext_vector_type(8)));
typedef _Float16 h4   __attribute__((ext_vector_type(4)));
typedef float    v8f  __attribute__((ext_vector_type(8)));
typedef unsigned int u32x4 __attribute__((ext_vector_type(4)));
typedef int          i32x8 __attribute__((ext_vector_type(8)));
typedef int          i32x4 __attribute__((ext_vector_type(4)));

#define B_   8
#define S_   1024
#define D_   1024
#define H_   16
#define DH_  64
#define OD_  (H_*DH_)
#define NEGM 1.0e12f

#if __has_builtin(__builtin_amdgcn_tensor_load_to_lds)
#define HAVE_TDM 1
#else
#define HAVE_TDM 0
#endif

static __device__ __forceinline__ v16h pack16(h8 lo, h8 hi) {
  v16h r;
#pragma unroll
  for (int i = 0; i < 8; ++i) { r[i] = lo[i]; r[i + 8] = hi[i]; }
  return r;
}

// ---------------------------------------------------------------------------
// Kernel 1: P = X @ W  (f32 in -> f16 out, head-major [B][H][S][DH])
// 128x128 tile per WG, K-step 32, 8 waves: wave = (wm 0..3) x (wn 0..1),
// each wave computes 32x64 = 2x4 WMMA 16x16 accumulators.
// ---------------------------------------------------------------------------
#define TM 128
#define TN 128
#define TK 32
#define LDA 40   // padded LDS row stride in halves (80B, 16B-aligned)

__global__ __launch_bounds__(256) void proj_kernel(
    const float* __restrict__ Xq, const float* __restrict__ Xk, const float* __restrict__ Xv,
    const float* __restrict__ Wq, const float* __restrict__ Wk, const float* __restrict__ Wv,
    _Float16* __restrict__ Oq, _Float16* __restrict__ Ok, _Float16* __restrict__ Ov)
{
  __shared__ _Float16 As[TM * LDA];   // X tile, row-major [m][k]
  __shared__ _Float16 Bs[TN * LDA];   // W tile, transposed [n][k]

  const int z = blockIdx.z;
  const float* X = (z == 0) ? Xq : ((z == 1) ? Xk : Xv);
  const float* W = (z == 0) ? Wq : ((z == 1) ? Wk : Wv);
  _Float16*   Ot = (z == 0) ? Oq : ((z == 1) ? Ok : Ov);

  const int tid = threadIdx.x;
  const int n0 = blockIdx.x * TN;
  const int m0 = blockIdx.y * TM;

  const int wave = tid >> 5, lane = tid & 31;
  const int wm = wave & 3, wn = wave >> 2;
  const int lr = lane & 15, g = lane >> 4;

  v8f zero8 = {0, 0, 0, 0, 0, 0, 0, 0};
  v8f acc[2][4];
#pragma unroll
  for (int i = 0; i < 2; ++i)
#pragma unroll
    for (int j = 0; j < 4; ++j) acc[i][j] = zero8;

  for (int k0 = 0; k0 < D_; k0 += TK) {
    __syncthreads();
#pragma unroll
    for (int i = 0; i < 4; ++i) {
      int idx = tid + i * 256;                 // 0..1023
      // --- X tile: 128 rows x 32 k-floats ---
      int row = idx >> 3;
      int c4  = (idx & 7) << 2;
      float4 xv = *(const float4*)&X[(size_t)(m0 + row) * D_ + k0 + c4];
      h4 hx = { (_Float16)xv.x, (_Float16)xv.y, (_Float16)xv.z, (_Float16)xv.w };
      *(h4*)&As[row * LDA + c4] = hx;
      // --- W tile: 32 k-rows x 128 n-floats, stored transposed ---
      int krow = idx >> 5;
      int nc   = (idx & 31) << 2;
      float4 wv = *(const float4*)&W[(size_t)(k0 + krow) * OD_ + n0 + nc];
      Bs[(nc + 0) * LDA + krow] = (_Float16)wv.x;
      Bs[(nc + 1) * LDA + krow] = (_Float16)wv.y;
      Bs[(nc + 2) * LDA + krow] = (_Float16)wv.z;
      Bs[(nc + 3) * LDA + krow] = (_Float16)wv.w;
      if (k0 + TK < D_) {
        __builtin_prefetch(&X[(size_t)(m0 + row) * D_ + k0 + TK + c4], 0, 1);
        __builtin_prefetch(&W[(size_t)(k0 + TK + krow) * OD_ + n0 + nc], 0, 1);
      }
    }
    __syncthreads();

    v16h af[2], bf[4];
#pragma unroll
    for (int ti = 0; ti < 2; ++ti) {
      int off = (wm * 32 + ti * 16 + lr) * LDA + g * 8;
      af[ti] = pack16(*(const h8*)&As[off], *(const h8*)&As[off + 16]);
    }
#pragma unroll
    for (int tj = 0; tj < 4; ++tj) {
      int off = (wn * 64 + tj * 16 + lr) * LDA + g * 8;
      bf[tj] = pack16(*(const h8*)&Bs[off], *(const h8*)&Bs[off + 16]);
    }
#pragma unroll
    for (int ti = 0; ti < 2; ++ti)
#pragma unroll
      for (int tj = 0; tj < 4; ++tj)
        acc[ti][tj] = __builtin_amdgcn_wmma_f32_16x16x32_f16(
            false, af[ti], false, bf[tj], (short)0, acc[ti][tj], false, false);
  }

  // Epilogue: f32 accum -> f16, scatter into [B][H][S][DH]
#pragma unroll
  for (int ti = 0; ti < 2; ++ti)
#pragma unroll
    for (int tj = 0; tj < 4; ++tj)
#pragma unroll
      for (int r = 0; r < 8; ++r) {
        int m = m0 + wm * 32 + ti * 16 + r + g * 8;   // global row (b*S + s)
        int n = n0 + wn * 64 + tj * 16 + lr;          // global col (h*DH + dh)
        int b  = m >> 10, s  = m & (S_ - 1);
        int hh = n >> 6,  dh = n & (DH_ - 1);
        Ot[(((size_t)(b * H_ + hh) * S_ + s) * DH_) + dh] = (_Float16)acc[ti][tj][r];
      }
}

// ---------------------------------------------------------------------------
// Kernel 2: flash attention per (b, h, 128-query tile). 8 waves x 16 queries.
// 32-key chunks; QK^T = 4 WMMA, PV = 4 WMMA per chunk per wave.
// K tile staged by the Tensor Data Mover (TDM) with LDS padding matching LDK.
// ---------------------------------------------------------------------------
#define LDK 72   // K chunk row stride  (144B, 16B-aligned) == 64 + TDM pad of 4 DW
#define LDV 40   // V^T row stride      (80B, 16B-aligned)
#define LDP 40   // per-wave P row stride

__global__ __launch_bounds__(256) void attn_kernel(
    const _Float16* __restrict__ qb, const _Float16* __restrict__ kb,
    const _Float16* __restrict__ vb,
    const int* __restrict__ Qlen, const int* __restrict__ Vlen,
    float* __restrict__ out)
{
  __shared__ _Float16 Ks[32 * LDK];        // [key][dh]  (TDM destination)
  __shared__ _Float16 Vt[DH_ * LDV];       // [dh][key] (transposed)
  __shared__ _Float16 Ps[8 * 16 * LDP];    // per-wave P scratch

  const int tid = threadIdx.x;
  const int wave = tid >> 5, lane = tid & 31;
  const int lr = lane & 15, g = lane >> 4;
  const int qt = blockIdx.x, hh = blockIdx.y, b = blockIdx.z;
  const size_t headBase = (size_t)(b * H_ + hh) * S_ * DH_;
  const int q0 = qt * 128 + wave * 16;
  const int Vl = Vlen[b];
  const int Ql = Qlen[b];

  // Q fragments (16 queries x 64 dh = two 16x32 A fragments)
  v16h qf[2];
#pragma unroll
  for (int c = 0; c < 2; ++c) {
    size_t base = headBase + (size_t)(q0 + lr) * DH_ + c * 32 + g * 8;
    qf[c] = pack16(*(const h8*)&qb[base], *(const h8*)&qb[base + 16]);
  }

  v8f zero8 = {0, 0, 0, 0, 0, 0, 0, 0};
  float Mr[8], Lr[8];
  v8f O[4];
#pragma unroll
  for (int r = 0; r < 8; ++r) { Mr[r] = -3.0e38f; Lr[r] = 0.0f; }
#pragma unroll
  for (int n = 0; n < 4; ++n) O[n] = zero8;

  const int kend = (qt + 1) * 128;         // exact causal bound for this tile
  const int key = tid >> 3;                // 0..31
  const int dh8 = (tid & 7) << 3;          // 0,8,..,56

#if HAVE_TDM
  // TDM descriptor skeleton for one 32x64 f16 tile, LDS-padded to 72-half rows:
  // row = 64 halves = 32 DWORDs -> pad_interval=4 (every 32 DW), pad_amount=3 (4 DW)
  const unsigned int ldsK = (unsigned int)(unsigned long long)(void*)&Ks[0];
  i32x8 tg1;
  tg1[0] = (1 << 16) | (1 << 20) | (4 << 22) | (3 << 25); // data_size=2B, pad_en, intv, amt
  tg1[1] = (int)(64u << 16);   // tensor_dim0 = 64 (low 16 in [31:16])
  tg1[2] = (int)(32u << 16);   // tensor_dim0 hi = 0 ; tensor_dim1 = 32 (low 16 in [31:16])
  tg1[3] = (int)(64u << 16);   // tensor_dim1 hi = 0 ; tile_dim0 = 64
  tg1[4] = 32;                 // tile_dim1 = 32 ; tile_dim2 = 0
  tg1[5] = 64;                 // tensor_dim0_stride = 64 (lo 32)
  tg1[6] = 0;
  tg1[7] = 0;
  const i32x4 tgz4 = {0, 0, 0, 0};
  const i32x8 tgz8 = {0, 0, 0, 0, 0, 0, 0, 0};
#endif

  for (int kc = 0; kc < kend; kc += 32) {
    __syncthreads();
#if HAVE_TDM
    if (wave == 0) {
      unsigned long long ga =
          (unsigned long long)(const void*)(kb + headBase + (size_t)kc * DH_);
      u32x4 tg0;
      tg0[0] = 1u;                                      // count=1, user descriptor
      tg0[1] = ldsK;                                    // LDS destination
      tg0[2] = (unsigned int)(ga & 0xFFFFFFFFu);        // global_addr[31:0]
      tg0[3] = (unsigned int)((ga >> 32) & 0x1FFFFFFu)  // global_addr[56:32]
               | (2u << 30);                            // type=2 (image/TDM)
      __builtin_amdgcn_tensor_load_to_lds(tg0, tg1, tgz4, tgz4, tgz8, 0);
    }
#endif
    {
      size_t gbase = headBase + (size_t)(kc + key) * DH_ + dh8;
#if !HAVE_TDM
      h8 kvv = *(const h8*)&kb[gbase];
      *(h8*)&Ks[key * LDK + dh8] = kvv;
#endif
      h8 vvv = *(const h8*)&vb[gbase];
#pragma unroll
      for (int j = 0; j < 8; ++j) Vt[(dh8 + j) * LDV + key] = vvv[j];
      if (kc + 32 < kend) __builtin_prefetch(&vb[gbase + (size_t)32 * DH_], 0, 1);
    }
#if HAVE_TDM
    if (wave == 0) __builtin_amdgcn_s_wait_tensorcnt(0);
#endif
    __syncthreads();

    // ---- scores: S(16q x 32k) = Q @ K^T, two 16x16 tiles, k-dim 64 = 2x32 ----
    v8f sc[2] = {zero8, zero8};
#pragma unroll
    for (int t = 0; t < 2; ++t) {
      int kro = (t * 16 + lr) * LDK;
#pragma unroll
      for (int c = 0; c < 2; ++c) {
        int off = kro + c * 32 + g * 8;
        v16h kf = pack16(*(const h8*)&Ks[off], *(const h8*)&Ks[off + 16]);
        sc[t] = __builtin_amdgcn_wmma_f32_16x16x32_f16(
            false, qf[c], false, kf, (short)0, sc[t], false, false);
      }
    }

    // ---- scale + additive masks (matches reference -1e12 semantics) ----
    const float scale = 0.125f;  // 1/sqrt(64)
#pragma unroll
    for (int t = 0; t < 2; ++t) {
      int kk = kc + t * 16 + lr;
      float lenpen = (kk >= Vl) ? NEGM : 0.0f;
#pragma unroll
      for (int r = 0; r < 8; ++r) {
        int qq = q0 + r + g * 8;
        float s = sc[t][r] * scale - lenpen;
        if (kk > qq) s -= NEGM;
        sc[t][r] = s;
      }
    }

    // ---- online softmax; rows live in 16-lane halves, reduce via shfl_xor ----
    float corr[8];
#pragma unroll
    for (int r = 0; r < 8; ++r) {
      float m2 = fmaxf(sc[0][r], sc[1][r]);
      m2 = fmaxf(m2, __shfl_xor(m2, 1, 32));
      m2 = fmaxf(m2, __shfl_xor(m2, 2, 32));
      m2 = fmaxf(m2, __shfl_xor(m2, 4, 32));
      m2 = fmaxf(m2, __shfl_xor(m2, 8, 32));
      float nm = fmaxf(Mr[r], m2);
      corr[r] = __expf(Mr[r] - nm);
      Mr[r] = nm;
      float p0 = __expf(sc[0][r] - nm);
      float p1 = __expf(sc[1][r] - nm);
      sc[0][r] = p0; sc[1][r] = p1;
      float s2 = p0 + p1;
      s2 += __shfl_xor(s2, 1, 32);
      s2 += __shfl_xor(s2, 2, 32);
      s2 += __shfl_xor(s2, 4, 32);
      s2 += __shfl_xor(s2, 8, 32);
      Lr[r] = Lr[r] * corr[r] + s2;
    }
#pragma unroll
    for (int n = 0; n < 4; ++n)
#pragma unroll
      for (int r = 0; r < 8; ++r) O[n][r] *= corr[r];

    // ---- repack P (C/D layout) -> A-fragment layout via per-wave LDS ----
    _Float16* pw = &Ps[wave * 16 * LDP];
#pragma unroll
    for (int t = 0; t < 2; ++t)
#pragma unroll
      for (int r = 0; r < 8; ++r)
        pw[(r + g * 8) * LDP + t * 16 + lr] = (_Float16)sc[t][r];
    v16h pf = pack16(*(const h8*)&pw[lr * LDP + g * 8],
                     *(const h8*)&pw[lr * LDP + 16 + g * 8]);

    // ---- O += P @ V : 4 dh-tiles of 16, k-dim 32 ----
#pragma unroll
    for (int n = 0; n < 4; ++n) {
      int off = (n * 16 + lr) * LDV + g * 8;
      v16h vf = pack16(*(const h8*)&Vt[off], *(const h8*)&Vt[off + 16]);
      O[n] = __builtin_amdgcn_wmma_f32_16x16x32_f16(
          false, pf, false, vf, (short)0, O[n], false, false);
    }
  }

  // ---- normalize, apply query-length mask, store f32 (B,S,H*DH) ----
#pragma unroll
  for (int r = 0; r < 8; ++r) {
    int qq = q0 + r + g * 8;
    float w = (qq < Ql) ? (1.0f / Lr[r]) : 0.0f;
#pragma unroll
    for (int n = 0; n < 4; ++n) {
      int dh = n * 16 + lr;
      out[((size_t)b * S_ + qq) * OD_ + hh * DH_ + dh] = O[n][r] * w;
    }
  }
}

// ---------------------------------------------------------------------------
extern "C" void kernel_launch(void* const* d_in, const int* in_sizes, int n_in,
                              void* d_out, int out_size, void* d_ws, size_t ws_size,
                              hipStream_t stream)
{
  (void)in_sizes; (void)n_in; (void)out_size; (void)ws_size;
  const float* Qs = (const float*)d_in[0];
  const float* Kse = (const float*)d_in[1];
  const float* Vs = (const float*)d_in[2];
  const int*   Ql = (const int*)d_in[3];
  const int*   Vl = (const int*)d_in[4];
  const float* WQ = (const float*)d_in[5];
  const float* WK = (const float*)d_in[6];
  const float* WV = (const float*)d_in[7];
  float* out = (float*)d_out;

  const size_t per = (size_t)B_ * S_ * H_ * DH_;   // 8M halves = 16MB
  _Float16* qh = (_Float16*)d_ws;
  _Float16* kh = qh + per;
  _Float16* vh = kh + per;

  dim3 g1(OD_ / TN, (B_ * S_) / TM, 3), b1(256);
  proj_kernel<<<g1, b1, 0, stream>>>(Qs, Kse, Vs, WQ, WK, WV, qh, kh, vh);

  dim3 g2(S_ / 128, H_, B_), b2(256);
  attn_kernel<<<g2, b2, 0, stream>>>(qh, kh, vh, Ql, Vl, out);
}
